// RNN_2095944040491
// MI455X (gfx1250) — matrix operand order repaired and checked by reading, not compiled, not run on hardware
//
#include <hip/hip_runtime.h>
#include <hip/hip_bf16.h>
#include <math.h>

// Problem sizes (match reference)
#define B_ 64
#define T_ 512
#define I_ 128
#define H_ 1024
#define O_ 128
#define ALPHA_ 0.2f

#define TPB 512            // 16 waves of 32
#define NWAVE (TPB / 32)

#if __has_builtin(__builtin_amdgcn_global_load_async_to_lds_b128)
#define HAVE_ASYNC_LDS 1
#endif

// ---------------------------------------------------------------------------
// WMMA types / helpers (CDNA5 gfx1250, wave32)
// ---------------------------------------------------------------------------
typedef __attribute__((ext_vector_type(16))) __bf16 v16bf;
typedef __attribute__((ext_vector_type(8)))  __bf16 v8bf;
typedef __attribute__((ext_vector_type(8)))  float  v8f;
typedef int v4i __attribute__((vector_size(4 * sizeof(int))));

// Explicit global-address-space views so weight streams lower to
// global_load_b128 (LOADcnt only) instead of flat_load_b128 (LOADcnt+DScnt,
// ties up the LDS datapath).
typedef __attribute__((address_space(1))) const __bf16 gcbf16;
typedef __attribute__((address_space(1))) const v8bf  gcv8bf;

static __device__ inline v16bf combine16(v8bf lo, v8bf hi) {
  return __builtin_shufflevector(lo, hi, 0, 1, 2, 3, 4, 5, 6, 7,
                                         8, 9, 10, 11, 12, 13, 14, 15);
}

// A-fragment (16x32 bf16): lane<16 holds K {k0..k0+15}, k0=kt*32;
// lanes>=16 the same with k0+16 (caller passes the shifted k0).
static __device__ inline v16bf loadA_frag(const __bf16* rowbase, int k0) {
  v8bf lo = *(const v8bf*)(rowbase + k0);
  v8bf hi = *(const v8bf*)(rowbase + k0 + 8);
  return combine16(lo, hi);
}

// B-fragment (32x16 bf16) from a pre-swizzled 512-element block in global
// memory: lane l owns 16 contiguous bf16 (32 bytes) at block + l*16.
static __device__ inline v16bf loadB_fragG(gcbf16* block, int lane) {
  gcv8bf* p = (gcv8bf*)(block + lane * 16);
  return combine16(p[0], p[1]);
}

static __device__ inline v8f wmma_bf16(v16bf a, v16bf b, v8f c) {
  return __builtin_amdgcn_wmma_f32_16x16x32_bf16(false, a, false, b,
                                                 (short)0, c, false, false);
}

// ---------------------------------------------------------------------------
// Prep kernels: fold g into weights, transpose Wrec, swizzle into B-fragment
// order.  Packed block id for (ktile, ntile) is (nt * KT + kt); inside a block
// lane = nin + 16*(kin>=16), element j = kin & 15.
// ---------------------------------------------------------------------------
__global__ void prep_wrec(const float* __restrict__ wrec,
                          const float* __restrict__ g,
                          __bf16* __restrict__ out) {
  int idx = blockIdx.x * blockDim.x + threadIdx.x;
  if (idx >= H_ * H_) return;
  int k = idx / H_, n = idx % H_;                 // W'[k][n] = g[k]*wrec[n][k]
  float v = g[k] * wrec[n * H_ + k];
  int kt = k >> 5, kin = k & 31, nt = n >> 4, nin = n & 15;
  int lane = (kin < 16) ? nin : (nin + 16);
  int j = kin & 15;
  out[(size_t)(nt * (H_ / 32) + kt) * 512 + lane * 16 + j] = (__bf16)v;
}

__global__ void prep_wi(const float* __restrict__ wi,
                        __bf16* __restrict__ out) {
  int idx = blockIdx.x * blockDim.x + threadIdx.x;
  if (idx >= I_ * H_) return;
  int k = idx / H_, n = idx % H_;                 // W'[k][n] = wi[k][n]
  float v = wi[k * H_ + n];
  int kt = k >> 5, kin = k & 31, nt = n >> 4, nin = n & 15;
  int lane = (kin < 16) ? nin : (nin + 16);
  int j = kin & 15;
  out[(size_t)(nt * (I_ / 32) + kt) * 512 + lane * 16 + j] = (__bf16)v;
}

__global__ void prep_wout(const float* __restrict__ wout,
                          const float* __restrict__ g,
                          __bf16* __restrict__ out) {
  int idx = blockIdx.x * blockDim.x + threadIdx.x;
  if (idx >= H_ * O_) return;
  int k = idx / O_, n = idx % O_;                 // W'[k][n] = g[k]*wout[k][n]
  float v = g[k] * wout[k * O_ + n];
  int kt = k >> 5, kin = k & 31, nt = n >> 4, nin = n & 15;
  int lane = (kin < 16) ? nin : (nin + 16);
  int j = kin & 15;
  out[(size_t)(nt * (H_ / 32) + kt) * 512 + lane * 16 + j] = (__bf16)v;
}

__global__ void prep_x(const float* __restrict__ x, __bf16* __restrict__ xbf,
                       int n) {
  for (int i = blockIdx.x * blockDim.x + threadIdx.x; i < n;
       i += gridDim.x * blockDim.x)
    xbf[i] = (__bf16)x[i];
}

// t = 0 output row: identical for every batch row.
__global__ void out0_kernel(const float* __restrict__ g,
                            const float* __restrict__ b,
                            const float* __restrict__ h0,
                            const float* __restrict__ wout,
                            float* __restrict__ out) {
  __shared__ float rg[H_];
  int tid = threadIdx.x;                          // 128 threads
  for (int k = tid; k < H_; k += 128)
    rg[k] = g[k] * tanhf(h0[k] + b[k]);
  __syncthreads();
  float s = 0.0f;
  for (int k = 0; k < H_; ++k) s += rg[k] * wout[k * O_ + tid];
  for (int bb = 0; bb < B_; ++bb)
    out[(size_t)bb * T_ * O_ + tid] = s;
}

// ---------------------------------------------------------------------------
// Main persistent RNN kernel: 4 workgroups (one per 16-row batch tile),
// 16 waves each.  Wave w owns H-columns [w*64, w*64+64) (4 WMMA N-tiles):
// its h state lives in 4 v8f accumulator slabs, r_t lives in LDS, the x_t
// tile is async-DMA'd into a double-buffered LDS staging area, and the
// 16x128 output tile is reduced across waves with ds_add_f32.
// ---------------------------------------------------------------------------
__global__ __launch_bounds__(TPB) void rnn_main(
    const __bf16* __restrict__ xbf,
    const __bf16* __restrict__ wrec_bf,
    const __bf16* __restrict__ wi_bf,
    const __bf16* __restrict__ wout_bf,
    const float* __restrict__ b,
    const float* __restrict__ h0,
    float* __restrict__ out) {
  __shared__ __bf16 rA[16 * H_];        // r_t, row-major [m][k]     (32 KB)
  __shared__ __bf16 xbuf[2][16 * I_];   // x_t staging, double-buf  (2x4 KB)
  __shared__ float outAcc[16 * O_];     // out tile reduction         (8 KB)

  const int tid = threadIdx.x;
  const int lane = tid & 31;
  const int w = tid >> 5;
  const int bt = blockIdx.x;            // batch tile (16 rows)
  const int ln16 = lane & 15;
  const int hiHalf = lane >> 4;
  const int n0 = w * 4;                 // first of 4 owned N-tiles

  float h[4][8];
  float bv[4];
#pragma unroll
  for (int j = 0; j < 4; ++j) {
    int col = (n0 + j) * 16 + ln16;
    bv[j] = b[col];
    float h0v = h0[col];
#pragma unroll
    for (int v = 0; v < 8; ++v) h[j][v] = h0v;
  }

  // r0 = tanh(h0 + b), identical across the 16 rows.
  for (int k = tid; k < H_; k += TPB) {
    __bf16 bb0 = (__bf16)tanhf(h0[k] + b[k]);
#pragma unroll
    for (int m = 0; m < 16; ++m) rA[m * H_ + k] = bb0;
  }

  // ---- stage x tile for t=0 into xbuf[0] ----
#ifdef HAVE_ASYNC_LDS
  if (w == 0) {
    int row = lane >> 1;
#pragma unroll
    for (int i = 0; i < 8; ++i) {
      int ch = i * 2 + (lane & 1);                 // 16-byte chunk in row
      const __bf16* gp =
          xbf + (size_t)(bt * 16 + row) * T_ * I_ + 0 * I_ + ch * 8;
      __bf16* lp = &xbuf[0][row * I_ + ch * 8];
      __builtin_amdgcn_global_load_async_to_lds_b128((v4i*)gp, (v4i*)lp, 0, 0);
    }
    asm volatile("s_wait_asynccnt 0x0" ::: "memory");
  }
#else
#pragma unroll
  for (int j = 0; j < (16 * I_) / TPB; ++j) {
    int i = tid + j * TPB;
    int row = i >> 7, e = i & 127;
    xbuf[0][i] = xbf[(size_t)(bt * 16 + row) * T_ * I_ + e];
  }
#endif
  __syncthreads();

  const int rowA = ln16;                   // this lane's A-matrix row
  const int kshift = hiHalf ? 16 : 0;      // upper half-lanes: K base +16

  // Opaque copies of the weight pointers; the asm barrier inside the loop
  // stops the compiler from hoisting t-invariant B-fragments into registers
  // (which caused scratch spills).  They are cast back to address_space(1)
  // pointers so the loads stay global_load (not flat_load).
  unsigned long wrec_u = (unsigned long)wrec_bf;
  unsigned long wi_u = (unsigned long)wi_bf;
  unsigned long wout_u = (unsigned long)wout_bf;

  for (int t = 0; t < T_ - 1; ++t) {
    const int cur = t & 1, nxt = cur ^ 1;
    asm("" : "+s"(wrec_u), "+s"(wi_u), "+s"(wout_u));
    gcbf16* wrec_t = (gcbf16*)wrec_u;
    gcbf16* wi_t = (gcbf16*)wi_u;
    gcbf16* wout_t = (gcbf16*)wout_u;

    // ---- prefetch x_{t+1} tile into xbuf[nxt] (async DMA, overlaps WMMA) --
#ifdef HAVE_ASYNC_LDS
    if (w == 0) {
      int row = lane >> 1;
#pragma unroll
      for (int i = 0; i < 8; ++i) {
        int ch = i * 2 + (lane & 1);
        const __bf16* gp = xbf + (size_t)(bt * 16 + row) * T_ * I_ +
                           (size_t)(t + 1) * I_ + ch * 8;
        __bf16* lp = &xbuf[nxt][row * I_ + ch * 8];
        __builtin_amdgcn_global_load_async_to_lds_b128((v4i*)gp, (v4i*)lp, 0,
                                                       0);
      }
    }
#else
#pragma unroll
    for (int j = 0; j < (16 * I_) / TPB; ++j) {
      int i = tid + j * TPB;
      int row = i >> 7, e = i & 127;
      xbuf[nxt][i] =
          xbf[(size_t)(bt * 16 + row) * T_ * I_ + (size_t)(t + 1) * I_ + e];
    }
#endif

    // ---- phase 1: S = r_t @ W'rec + x_t @ W'i (per-wave 16x64 slab) ----
    v8f c[4] = {{}, {}, {}, {}};
#pragma unroll 2
    for (int kt = 0; kt < H_ / 32; ++kt) {
      v16bf a = loadA_frag(&rA[rowA * H_], kt * 32 + kshift);
#pragma unroll
      for (int j = 0; j < 4; ++j) {
        v16bf bf = loadB_fragG(
            wrec_t + (size_t)((n0 + j) * (H_ / 32) + kt) * 512, lane);
        c[j] = wmma_bf16(a, bf, c[j]);
      }
    }
#pragma unroll
    for (int kt = 0; kt < I_ / 32; ++kt) {
      v16bf a = loadA_frag(&xbuf[cur][rowA * I_], kt * 32 + kshift);
#pragma unroll
      for (int j = 0; j < 4; ++j) {
        v16bf bf = loadB_fragG(
            wi_t + (size_t)((n0 + j) * (I_ / 32) + kt) * 512, lane);
        c[j] = wmma_bf16(a, bf, c[j]);
      }
    }

    // h <- (1-a)h + a*S ; r <- tanh(h + b)
    float rn[4][8];
#pragma unroll
    for (int j = 0; j < 4; ++j)
#pragma unroll
      for (int v = 0; v < 8; ++v) {
        h[j][v] = (1.0f - ALPHA_) * h[j][v] + ALPHA_ * c[j][v];
        rn[j][v] = tanhf(h[j][v] + bv[j]);
      }

    __syncthreads();  // all waves finished reading rA(old) and xbuf[cur-1]
#pragma unroll
    for (int j = 0; j < 4; ++j) {
      int col = (n0 + j) * 16 + ln16;
#pragma unroll
      for (int v = 0; v < 8; ++v) {
        int row = v + hiHalf * 8;          // C-matrix M mapping
        rA[row * H_ + col] = (__bf16)rn[j][v];
      }
    }
#pragma unroll
    for (int j = 0; j < (16 * O_) / TPB; ++j) outAcc[tid + j * TPB] = 0.0f;
    __syncthreads();

    // ---- phase 2: out_{t+1} = r_{t+1} @ W'out (K split 2-way per N tile) --
    const int nt8 = w & 7;                 // output column tile (of 8)
    const int q = w >> 3;                  // K half (of 2)
    v8f co = {};
#pragma unroll 4
    for (int kk = 0; kk < 16; ++kk) {
      int kt = q * 16 + kk;
      v16bf a = loadA_frag(&rA[rowA * H_], kt * 32 + kshift);
      v16bf bo =
          loadB_fragG(wout_t + (size_t)(nt8 * (H_ / 32) + kt) * 512, lane);
      co = wmma_bf16(a, bo, co);
    }
    const int colO = nt8 * 16 + ln16;
#pragma unroll
    for (int v = 0; v < 8; ++v) {
      int row = v + hiHalf * 8;
      atomicAdd(&outAcc[row * O_ + colO], co[v]);  // ds_add_f32
    }
#ifdef HAVE_ASYNC_LDS
    if (w == 0) asm volatile("s_wait_asynccnt 0x0" ::: "memory");
#endif
    __syncthreads();

    // store 16x128 f32 tile to out[:, t+1, :]
    {
      float* orow = out + (size_t)(bt * 16) * T_ * O_ + (size_t)(t + 1) * O_;
#pragma unroll
      for (int j = 0; j < (16 * O_) / TPB; ++j) {
        int i = tid + j * TPB;
        int m = i >> 7, o = i & 127;
        orow[(size_t)m * T_ * O_ + o] = outAcc[m * O_ + o];
      }
    }
  }
}

// ---------------------------------------------------------------------------
// Launch
// ---------------------------------------------------------------------------
extern "C" void kernel_launch(void* const* d_in, const int* in_sizes, int n_in,
                              void* d_out, int out_size, void* d_ws,
                              size_t ws_size, hipStream_t stream) {
  const float* x    = (const float*)d_in[0];
  const float* wi   = (const float*)d_in[1];
  const float* wrec = (const float*)d_in[2];
  const float* wout = (const float*)d_in[3];
  const float* b    = (const float*)d_in[4];
  const float* g    = (const float*)d_in[5];
  const float* h0   = (const float*)d_in[6];
  float* out = (float*)d_out;

  // Workspace layout (bf16 elements)
  __bf16* wsb     = (__bf16*)d_ws;
  __bf16* wrec_bf = wsb;                                // H*H
  __bf16* wi_bf   = wrec_bf + (size_t)H_ * H_;          // I*H
  __bf16* wout_bf = wi_bf + (size_t)I_ * H_;            // H*O
  __bf16* xbf     = wout_bf + (size_t)H_ * O_;          // B*T*I

  prep_wrec<<<(H_ * H_ + 255) / 256, 256, 0, stream>>>(wrec, g, wrec_bf);
  prep_wi<<<(I_ * H_ + 255) / 256, 256, 0, stream>>>(wi, wi_bf);
  prep_wout<<<(H_ * O_ + 255) / 256, 256, 0, stream>>>(wout, g, wout_bf);
  prep_x<<<4096, 256, 0, stream>>>(x, xbf, B_ * T_ * I_);
  out0_kernel<<<1, 128, 0, stream>>>(g, b, h0, wout, out);

  rnn_main<<<B_ / 16, TPB, 0, stream>>>(xbf, wrec_bf, wi_bf, wout_bf, b, h0,
                                        out);
}
